// TransformerBlock_36490042147635
// MI455X (gfx1250) — compile-verified, matrix operand order checked by reading, not run on hardware
//
#include <hip/hip_runtime.h>
#include <cstdint>
#include <cstddef>

// ---------------------------------------------------------------------------
// MI455X (gfx1250) implementation notes:
//  * wave32 only; all channel GEMMs -> v_wmma_f32_16x16x32_bf16.
//  * GEMM block = one 16-pixel n-tile x (M/16) waves. The B tile
//    (16 rows x K bf16) is staged global->LDS with the Tensor Data Mover
//    (TENSOR_LOAD_TO_LDS + s_wait_tensorcnt) when the builtin exists;
//    weights are converted fp32->bf16 once per block into LDS. Inner loop is
//    pure ds_load_b128 + v_wmma.
//  * Activations between GEMMs are pixel-major bf16 [pixel][C] so fragment
//    rows are contiguous (ISA 7.12.2 16-bit layouts).
//  * Workspace: ~496 MiB, lifetime-aliased regions (see offsets below).
// ---------------------------------------------------------------------------

typedef __bf16 bf16;
typedef bf16  v16bf __attribute__((ext_vector_type(16)));
typedef float v8f   __attribute__((ext_vector_type(8)));
typedef unsigned int u32x4 __attribute__((ext_vector_type(4)));
typedef int i32x4 __attribute__((ext_vector_type(4)));
typedef int i32x8 __attribute__((ext_vector_type(8)));

struct alignas(16) U128 { unsigned int w[4]; };
union  Frag  { v16bf v; bf16 h[16]; };
struct alignas(16) Half8 { bf16 h[8]; };

#define HWPIX 65536      // 256*256
#define NPIX  262144     // 4*65536

// ---- TDM availability / arity detection -----------------------------------
#if defined(__gfx1250__) && __has_builtin(__builtin_amdgcn_tensor_load_to_lds) && \
    __has_builtin(__builtin_amdgcn_s_wait_tensorcnt)
  #define HAS_TDM 1
  #if __has_include(<hip/amd_detail/amd_gfx1250_TDM.h>)
    // clang-23 / therock headers: 6-arg builtin
    #define TDM_LOAD(g0, g1, g2, g3) \
        __builtin_amdgcn_tensor_load_to_lds((g0), (g1), (g2), (g3), \
            (i32x8){0, 0, 0, 0, 0, 0, 0, 0}, 0)
  #else
    // ROCm 7.2 clang-22: 5-arg builtin
    #define TDM_LOAD(g0, g1, g2, g3) \
        __builtin_amdgcn_tensor_load_to_lds((g0), (g1), (g2), (g3), 0)
  #endif
#endif

__device__ __forceinline__ float sigf(float x) { return 1.0f / (1.0f + expf(-x)); }

__device__ const float FFT_CT[8] = {1.f, 0.70710678118f, 0.f, -0.70710678118f,
                                    -1.f, -0.70710678118f, 0.f, 0.70710678118f};
__device__ const float FFT_ST[8] = {0.f, 0.70710678118f, 1.f, 0.70710678118f,
                                    0.f, -0.70710678118f, -1.f, -0.70710678118f};

// ---------------------------------------------------------------------------
// WMMA GEMM over pixels:  OUT[m, p] = sum_k W[m,k] * Bm[p, bco+k]
//  grid.x = NPIX/16 (one 16-pixel tile per block), blockDim = 32*(M/16).
//  Wave w computes rows [16w, 16w+16). K multiple of 32, M multiple of 16.
// ---------------------------------------------------------------------------
__global__ void wmma_gemm_kernel(const bf16* __restrict__ Bm, int Bstride, int bco,
                                 const float* __restrict__ Wt,
                                 const float* __restrict__ bias,
                                 const float* __restrict__ res, int resCtot, int resCoff,
                                 float* __restrict__ outF, int outFCtot, int outFCoff,
                                 bf16* __restrict__ outB, int outBstride, int outBCoff,
                                 int M, int K)
{
    __shared__ alignas(16) bf16 wlds[16384];   // up to 256x64 bf16 weights (32KB)
    __shared__ alignas(16) bf16 blds[2048];    // 16 pixels x K<=128 bf16 (4KB)

    const int lane = threadIdx.x & 31;
    const int wave = threadIdx.x >> 5;
    const int mn   = lane & 15;       // A row / B,C,D column
    const int hi   = lane >> 4;       // half select per ISA 16-bit layouts
    const int m0   = wave << 4;
    const int P    = (blockIdx.x << 4) + mn;   // this lane's pixel

#if defined(HAS_TDM)
    // --- Stage B tile (16 rows x K bf16, row stride Bstride) via TDM -------
    if (wave == 0) {
        const unsigned long long ga = (unsigned long long)(uintptr_t)
            (Bm + (size_t)(blockIdx.x << 4) * Bstride + bco);
        const unsigned int ldsOff = (unsigned int)(uintptr_t)(&blds[0]);
        // D# group0: count=1 | lds_addr | global_addr | type=2
        u32x4 g0 = { 1u,
                     ldsOff,
                     (unsigned int)ga,
                     (unsigned int)((ga >> 32) & 0x01FFFFFFu) | 0x80000000u };
        // D# group1: data_size=2B; tensor dims large (no OOB clip);
        // tile = K x 16; dim0 stride = Bstride elements.
        const unsigned int td = 0x40000000u;   // huge tensor extent
        i32x8 g1 = { (int)(1u << 16),                       // mask=0, data_size=1(2B)
                     (int)((td & 0xFFFFu) << 16),           // barrier=0 | td0 lo
                     (int)((td >> 16) | ((td & 0xFFFFu) << 16)), // td0 hi | td1 lo
                     (int)((td >> 16) | ((unsigned)K << 16)),    // td1 hi | tile_dim0=K
                     16,                                    // tile_dim1=16, tile_dim2=0
                     Bstride,                               // dim0 stride lo
                     0, 0 };                                // stride hi | dim1 stride
        i32x4 gz = { 0, 0, 0, 0 };
        TDM_LOAD(g0, g1, gz, gz);
    }
#else
    // --- Fallback: cooperative vector copy of the B tile -------------------
    {
        const int cpr = K >> 3;                 // 16B chunks per row
        for (int i = threadIdx.x; i < 16 * cpr; i += blockDim.x) {
            const int row = i / cpr;
            const int col = (i - row * cpr) << 3;
            *(U128*)(void*)(blds + row * K + col) =
                *(const U128*)(const void*)
                    (Bm + (size_t)((blockIdx.x << 4) + row) * Bstride + bco + col);
        }
    }
#endif

    // --- Stage weights fp32 -> bf16 once per block (packed pairs) ----------
    {
        unsigned int* w32 = (unsigned int*)wlds;
        const int half_n = (M * K) >> 1;
        for (int i = threadIdx.x; i < half_n; i += blockDim.x) {
            union { bf16 h[2]; unsigned int u; } pk;
            pk.h[0] = (bf16)Wt[2 * i];
            pk.h[1] = (bf16)Wt[2 * i + 1];
            w32[i] = pk.u;
        }
    }

#if defined(HAS_TDM)
    if (wave == 0) __builtin_amdgcn_s_wait_tensorcnt(0);
#endif
    __syncthreads();

    // --- Main K loop: 4x ds_load_b128 + 1x v_wmma per 32-wide K chunk ------
    v8f acc = {0.f, 0.f, 0.f, 0.f, 0.f, 0.f, 0.f, 0.f};
    const bf16* arow = wlds + (size_t)(m0 + mn) * K;
    const bf16* brow = blds + (size_t)mn * K;
    for (int kc = 0; kc < K; kc += 32) {
        Frag a, b;
        // Both 16-bit fragment layouts: halves 0..7 at k=kc+hi*8, 8..15 at +16.
        *(U128*)(void*)&a.h[0] = *(const U128*)(const void*)(arow + kc + hi * 8);
        *(U128*)(void*)&a.h[8] = *(const U128*)(const void*)(arow + kc + 16 + hi * 8);
        *(U128*)(void*)&b.h[0] = *(const U128*)(const void*)(brow + kc + hi * 8);
        *(U128*)(void*)&b.h[8] = *(const U128*)(const void*)(brow + kc + 16 + hi * 8);
        acc = __builtin_amdgcn_wmma_f32_16x16x32_bf16(
                  false, a.v, false, b.v, (short)0, acc, false, false);
    }

    // --- Epilogue: C/D layout lane=column mn, vgpr r -> row m0+hi*8+r ------
    const int bidx = P >> 16;
    const int pl   = P & (HWPIX - 1);
    Half8 ob;
#pragma unroll
    for (int r = 0; r < 8; ++r) {
        const int m = m0 + hi * 8 + r;
        float v = acc[r];
        if (bias) v += bias[m];
        if (res)  v += res[((size_t)(bidx * resCtot + resCoff + m)) * HWPIX + pl];
        if (outF) outF[((size_t)(bidx * outFCtot + outFCoff + m)) * HWPIX + pl] = v;
        ob.h[r] = (bf16)v;
    }
    if (outB) {
        bf16* dst = outB + (size_t)P * outBstride + outBCoff + m0 + hi * 8;
        *(U128*)(void*)dst = *(U128*)(void*)&ob;
    }
}

// ---------------------------------------------------------------------------
// LayerNorm over 64 channels (NCHW fp32 in) -> pixel-major bf16 [p][64]
// ---------------------------------------------------------------------------
__global__ void ln_kernel(const float* __restrict__ x,
                          const float* __restrict__ w,
                          const float* __restrict__ b,
                          bf16* __restrict__ outT)
{
    const int P = blockIdx.x * blockDim.x + threadIdx.x;
    const int bidx = P >> 16, pl = P & (HWPIX - 1);
    const float* base = x + (size_t)bidx * 64 * HWPIX + pl;
    float s = 0.f, ss = 0.f;
    for (int c = 0; c < 64; ++c) {
        const float v = base[(size_t)c * HWPIX];
        s += v; ss += v * v;
    }
    const float mu  = s * (1.f / 64.f);
    const float var = ss * (1.f / 64.f) - mu * mu;
    const float inv = rsqrtf(var + 1e-5f);
    for (int c0 = 0; c0 < 64; c0 += 8) {
        Half8 ob;
#pragma unroll
        for (int i = 0; i < 8; ++i) {
            const int c = c0 + i;
            ob.h[i] = (bf16)((base[(size_t)c * HWPIX] - mu) * inv * w[c] + b[c]);
        }
        *(U128*)(void*)(outT + (size_t)P * 64 + c0) = *(U128*)(void*)&ob;
    }
}

// ---------------------------------------------------------------------------
// Depthwise conv helpers (SAME, cross-correlation like XLA)
// ---------------------------------------------------------------------------
__device__ __forceinline__ float conv3_at(const float* plane, int y, int x,
                                          const float* k)
{
    float acc = 0.f;
#pragma unroll
    for (int dy = -1; dy <= 1; ++dy) {
        const int yy = y + dy; if ((unsigned)yy > 255u) continue;
#pragma unroll
        for (int dx = -1; dx <= 1; ++dx) {
            const int xx = x + dx; if ((unsigned)xx > 255u) continue;
            acc += plane[yy * 256 + xx] * k[(dy + 1) * 3 + (dx + 1)];
        }
    }
    return acc;
}

__device__ __forceinline__ float conv5_at(const float* plane, int y, int x,
                                          const float* k)
{
    float acc = 0.f;
#pragma unroll
    for (int dy = -2; dy <= 2; ++dy) {
        const int yy = y + dy; if ((unsigned)yy > 255u) continue;
#pragma unroll
        for (int dx = -2; dx <= 2; ++dx) {
            const int xx = x + dx; if ((unsigned)xx > 255u) continue;
            acc += plane[yy * 256 + xx] * k[(dy + 2) * 5 + (dx + 2)];
        }
    }
    return acc;
}

__global__ void dwconv3_kernel(const float* __restrict__ in,
                               const float* __restrict__ kw,
                               float* __restrict__ out, int Ctot)
{
    const int c = blockIdx.y;
    const int P = blockIdx.x * 256 + threadIdx.x;
    const int bidx = P >> 16, pl = P & (HWPIX - 1), yy = pl >> 8, xx = pl & 255;
    const float* plane = in + (size_t)(bidx * Ctot + c) * HWPIX;
    out[(size_t)(bidx * Ctot + c) * HWPIX + pl] = conv3_at(plane, yy, xx, kw + c * 9);
}

// ---------------------------------------------------------------------------
// LGCE (CBAM-ish) pieces
// ---------------------------------------------------------------------------
__global__ void pool_kernel(const float* __restrict__ xl, float* __restrict__ pooled)
{
    const int bc = blockIdx.x;                 // b*32 + c
    const float* plane = xl + (size_t)bc * HWPIX;
    float s = 0.f;
    for (int i = threadIdx.x; i < HWPIX; i += 256) s += plane[i];
    __shared__ float red[256];
    red[threadIdx.x] = s; __syncthreads();
    for (int o = 128; o > 0; o >>= 1) {
        if (threadIdx.x < o) red[threadIdx.x] += red[threadIdx.x + o];
        __syncthreads();
    }
    if (threadIdx.x == 0) pooled[bc] = red[0] * (1.f / (float)HWPIX);
}

__global__ void cw_kernel(const float* __restrict__ pooled,
                          const float* __restrict__ m1w, const float* __restrict__ m1b,
                          const float* __restrict__ m2w, const float* __restrict__ m2b,
                          float* __restrict__ cw)
{
    __shared__ float h1[32];                   // [b][g]
    const int tid = threadIdx.x;
    if (tid < 32) {
        const int b_ = tid >> 3, g = tid & 7;
        float a = m1b[g];
        for (int c = 0; c < 4; ++c) a += pooled[b_ * 32 + g * 4 + c] * m1w[g * 4 + c];
        h1[tid] = fmaxf(a, 0.f);
    }
    __syncthreads();
    const int b_ = tid >> 5, g = (tid >> 2) & 7, c = tid & 3;
    cw[tid] = sigf(h1[b_ * 8 + g] * m2w[g * 4 + c] + m2b[g * 4 + c]);
}

__global__ void resp_kernel(const float* __restrict__ xl, const float* __restrict__ cw,
                            const float* __restrict__ spw, const float* __restrict__ spb,
                            float* __restrict__ resp, float* __restrict__ thrsum)
{
    const int g = blockIdx.y;
    const int P = blockIdx.x * 256 + threadIdx.x;
    const int bidx = P >> 16, pl = P & (HWPIX - 1);
    float xa[4];
    float acc = spb[g];
#pragma unroll
    for (int c = 0; c < 4; ++c) {
        const float v = xl[(size_t)(bidx * 32 + g * 4 + c) * HWPIX + pl];
        xa[c] = v * cw[bidx * 32 + g * 4 + c];
        acc += xa[c] * spw[g * 4 + c];
    }
    const float sw = sigf(acc);
    float ls = 0.f;
#pragma unroll
    for (int c = 0; c < 4; ++c) {
        const float r = sigf(xa[c] * sw);
        resp[(size_t)(bidx * 32 + g * 4 + c) * HWPIX + pl] = r;
        ls += r;
    }
#pragma unroll
    for (int o = 16; o > 0; o >>= 1) ls += __shfl_xor(ls, o, 32);
    if ((threadIdx.x & 31) == 0) atomicAdd(&thrsum[bidx * 8 + g], ls);
}

__global__ void mask_kernel(const float* __restrict__ xl, const float* __restrict__ resp,
                            const float* __restrict__ thrsum, bf16* __restrict__ maskedT)
{
    const int P = blockIdx.x * 256 + threadIdx.x;
    const int bidx = P >> 16, pl = P & (HWPIX - 1);
    for (int c0 = 0; c0 < 32; c0 += 8) {
        Half8 ob;
#pragma unroll
        for (int i = 0; i < 8; ++i) {
            const int c = c0 + i, g = c >> 2;
            const float thr = thrsum[bidx * 8 + g] * (1.f / (4.f * (float)HWPIX));
            const float r = resp[(size_t)(bidx * 32 + c) * HWPIX + pl];
            const float m = (r > thr) ? 1.f : r;
            ob.h[i] = (bf16)(xl[(size_t)(bidx * 32 + c) * HWPIX + pl] * m);
        }
        *(U128*)(void*)(maskedT + (size_t)P * 32 + c0) = *(U128*)(void*)&ob;
    }
}

// ---------------------------------------------------------------------------
// Gate mean: gmean accumulation for dk selection
// ---------------------------------------------------------------------------
__global__ void gate_mean_kernel(const bf16* __restrict__ projT,
                                 const float* __restrict__ w1, const float* __restrict__ b1,
                                 const float* __restrict__ w2, const float* __restrict__ b2,
                                 float* __restrict__ gsum)
{
    __shared__ float sw1[512], sb1[16], sw2[16], sb2;
    for (int i = threadIdx.x; i < 512; i += 256) sw1[i] = w1[i];
    if (threadIdx.x < 16) { sb1[threadIdx.x] = b1[threadIdx.x]; sw2[threadIdx.x] = w2[threadIdx.x]; }
    if (threadIdx.x == 0) sb2 = b2[0];
    __syncthreads();
    float local = 0.f;
    for (int P = blockIdx.x * 256 + threadIdx.x; P < NPIX; P += gridDim.x * 256) {
        const bf16* g = projT + (size_t)P * 64 + 32;
        __builtin_prefetch(g + (size_t)(gridDim.x * 256) * 64, 0, 1);
        float gl[32];
#pragma unroll
        for (int c = 0; c < 32; ++c) gl[c] = (float)g[c];
        float acc2 = sb2;
        for (int h = 0; h < 16; ++h) {
            float a = sb1[h];
#pragma unroll
            for (int c = 0; c < 32; ++c) a += sw1[h * 32 + c] * gl[c];
            acc2 += sw2[h] * fmaxf(a, 0.f);
        }
        local += sigf(acc2);
    }
#pragma unroll
    for (int o = 16; o > 0; o >>= 1) local += __shfl_xor(local, o, 32);
    if ((threadIdx.x & 31) == 0) atomicAdd(gsum, local);
}

// ---------------------------------------------------------------------------
// Channel-attention statistics: raw q.k dots + q/k sumsq (per b,h)
// stats layout per bh (80 floats): [64 dots c*8+d][8 qss][8 kss]
// ---------------------------------------------------------------------------
__global__ void qk_stats_kernel(const float* __restrict__ qkv, float* __restrict__ stats)
{
    const int bh = blockIdx.y;
    const int b_ = bh >> 2, h = bh & 3;
    const float* qb = qkv + (size_t)(b_ * 96 + h * 8) * HWPIX;
    const float* kb = qkv + (size_t)(b_ * 96 + 32 + h * 8) * HWPIX;
    float dot[64]; float qs[8]; float ks[8];
#pragma unroll
    for (int i = 0; i < 64; ++i) dot[i] = 0.f;
#pragma unroll
    for (int i = 0; i < 8; ++i) { qs[i] = 0.f; ks[i] = 0.f; }

    for (int pl = blockIdx.x * 256 + threadIdx.x; pl < HWPIX; pl += gridDim.x * 256) {
        float qv[8], kv[8];
#pragma unroll
        for (int i = 0; i < 8; ++i) {
            qv[i] = qb[(size_t)i * HWPIX + pl];
            kv[i] = kb[(size_t)i * HWPIX + pl];
        }
#pragma unroll
        for (int i = 0; i < 8; ++i) {
            qs[i] += qv[i] * qv[i];
            ks[i] += kv[i] * kv[i];
#pragma unroll
            for (int j = 0; j < 8; ++j) dot[i * 8 + j] += qv[i] * kv[j];
        }
    }
    float* st = stats + bh * 80;
#pragma unroll
    for (int i = 0; i < 64; ++i) {
        float v = dot[i];
#pragma unroll
        for (int o = 16; o > 0; o >>= 1) v += __shfl_xor(v, o, 32);
        if ((threadIdx.x & 31) == 0) atomicAdd(&st[i], v);
    }
#pragma unroll
    for (int i = 0; i < 8; ++i) {
        float v = qs[i];
#pragma unroll
        for (int o = 16; o > 0; o >>= 1) v += __shfl_xor(v, o, 32);
        if ((threadIdx.x & 31) == 0) atomicAdd(&st[64 + i], v);
    }
#pragma unroll
    for (int i = 0; i < 8; ++i) {
        float v = ks[i];
#pragma unroll
        for (int o = 16; o > 0; o >>= 1) v += __shfl_xor(v, o, 32);
        if ((threadIdx.x & 31) == 0) atomicAdd(&st[72 + i], v);
    }
}

// Finalize: normalize, temperature, top-dk mask, softmax, * sum(attn_scales)
__global__ void attn_finalize_kernel(const float* __restrict__ stats,
                                     const float* __restrict__ gsum,
                                     const float* __restrict__ temp,
                                     const float* __restrict__ scales,
                                     float* __restrict__ attnf)
{
    const int tid = threadIdx.x;           // 128 = 16 bh * 8 rows
    const float gmean = gsum[0] * (1.f / (float)NPIX);
    int dk = (int)floorf(8.f * gmean);
    dk = (dk < 1) ? 1 : ((dk > 8) ? 8 : dk);
    const float s = scales[0] + scales[1] + scales[2] + scales[3];
    const int bh = tid >> 3, c = tid & 7, h = bh & 3;
    const float* st = stats + bh * 80;
    const float qn = fmaxf(sqrtf(st[64 + c]), 1e-12f);
    float row[8], vals[8];
#pragma unroll
    for (int d = 0; d < 8; ++d) {
        const float kn = fmaxf(sqrtf(st[72 + d]), 1e-12f);
        row[d] = st[c * 8 + d] / (qn * kn) * temp[h];
        vals[d] = row[d];
    }
    for (int i = 0; i < dk; ++i) {          // partial selection sort (descending)
        int mi = i;
        for (int j = i + 1; j < 8; ++j) if (vals[j] > vals[mi]) mi = j;
        const float tv = vals[i]; vals[i] = vals[mi]; vals[mi] = tv;
    }
    const float kth = vals[dk - 1];
    float mx = -1e30f;
#pragma unroll
    for (int d = 0; d < 8; ++d) if (row[d] >= kth && row[d] > mx) mx = row[d];
    float e[8], sum = 0.f;
#pragma unroll
    for (int d = 0; d < 8; ++d) {
        e[d] = (row[d] >= kth) ? expf(row[d] - mx) : 0.f;
        sum += e[d];
    }
#pragma unroll
    for (int d = 0; d < 8; ++d) attnf[bh * 64 + c * 8 + d] = e[d] / sum * s;
}

// out[c, p] = sum_d attn[c,d] * v[d, p]  -> fusedT channels h*8+c (bf16)
__global__ void attn_apply_kernel(const float* __restrict__ qkv,
                                  const float* __restrict__ attnf,
                                  bf16* __restrict__ fusedT)
{
    const int bh = blockIdx.y;
    const int b_ = bh >> 2, h = bh & 3;
    __shared__ float A[64];
    if (threadIdx.x < 64) A[threadIdx.x] = attnf[bh * 64 + threadIdx.x];
    __syncthreads();
    const int pl = blockIdx.x * 256 + threadIdx.x;
    const float* vb = qkv + (size_t)(b_ * 96 + 64 + h * 8) * HWPIX;
    float vv[8];
#pragma unroll
    for (int d = 0; d < 8; ++d) vv[d] = vb[(size_t)d * HWPIX + pl];
    Half8 ob;
#pragma unroll
    for (int c = 0; c < 8; ++c) {
        float a = 0.f;
#pragma unroll
        for (int d = 0; d < 8; ++d) a += A[c * 8 + d] * vv[d];
        ob.h[c] = (bf16)a;
    }
    const int P = b_ * HWPIX + pl;
    *(U128*)(void*)(fusedT + (size_t)P * 64 + h * 8) = *(U128*)(void*)&ob;
}

// ---------------------------------------------------------------------------
// MFFN fused gated activation: y[ch] = gelu(dw3(t[ch])) * partner, bf16 [p][128]
// ---------------------------------------------------------------------------
__global__ void mffn_act_kernel(const float* __restrict__ t,
                                const float* __restrict__ dw3,
                                const float* __restrict__ dw5,
                                bf16* __restrict__ yT)
{
    const int ch = blockIdx.y;                  // 0..127
    const int P = blockIdx.x * 256 + threadIdx.x;
    const int bidx = P >> 16, pl = P & (HWPIX - 1), yy = pl >> 8, xx = pl & 255;
    const float* gplane = t + (size_t)(bidx * 256 + ch) * HWPIX;
    const float gate = conv3_at(gplane, yy, xx, dw3 + ch * 9);
    float part;
    if (ch < 64) {
        const float* pp = t + (size_t)(bidx * 256 + 128 + ch) * HWPIX;
        part = conv3_at(pp, yy, xx, dw3 + (128 + ch) * 9);
    } else {
        const float* pp = t + (size_t)(bidx * 256 + 192 + (ch - 64)) * HWPIX;
        part = conv5_at(pp, yy, xx, dw5 + (ch - 64) * 25);
    }
    const float g = 0.5f * gate * (1.f + erff(gate * 0.70710678118f));
    yT[(size_t)P * 128 + ch] = (bf16)(g * part);
}

// ---------------------------------------------------------------------------
// Per-8x8-patch real FFT filter (rfft2 * W -> irfft2) + residual, via LDS
// block = 256 threads = 4 patches x 64 threads (thread = (i,j) in patch)
// ---------------------------------------------------------------------------
__global__ void fft_kernel(const float* __restrict__ y2,
                           const float* __restrict__ xafter,
                           const float* __restrict__ fW,
                           float* __restrict__ out)
{
    __shared__ float Sre[4][8][8], Sim[4][8][8], Tre[4][8][8], Tim[4][8][8];
    __shared__ float Wf[4][8][5];
    const int tid = threadIdx.x;
    const int sp = tid >> 6;
    const int i = (tid >> 3) & 7, j = tid & 7;
    const int patch = blockIdx.x * 4 + sp;
    const int px = patch & 31, py = (patch >> 5) & 31;
    const int c = (patch >> 10) & 63, b_ = patch >> 16;
    const size_t plane = (size_t)(b_ * 64 + c) * HWPIX;
    const int yy = py * 8 + i, xx = px * 8 + j;

    const int tin = tid & 63;
    if (tin < 40) Wf[sp][tin / 5][tin % 5] = fW[(size_t)c * 40 + tin];
    Sre[sp][i][j] = y2[plane + yy * 256 + xx];
    __syncthreads();

    { // forward row DFT (v = j)
        float re = 0.f, im = 0.f;
#pragma unroll
        for (int jj = 0; jj < 8; ++jj) {
            const int k = (jj * j) & 7;
            re += Sre[sp][i][jj] * FFT_CT[k];
            im -= Sre[sp][i][jj] * FFT_ST[k];
        }
        Tre[sp][i][j] = re; Tim[sp][i][j] = im;
    }
    __syncthreads();
    { // forward col DFT (u = i) + real spectral multiply (Hermitian-extended W)
        float re = 0.f, im = 0.f;
#pragma unroll
        for (int ii = 0; ii < 8; ++ii) {
            const int k = (ii * i) & 7;
            re += Tre[sp][ii][j] * FFT_CT[k] + Tim[sp][ii][j] * FFT_ST[k];
            im += Tim[sp][ii][j] * FFT_CT[k] - Tre[sp][ii][j] * FFT_ST[k];
        }
        const float w = (j <= 4) ? Wf[sp][i][j] : Wf[sp][(8 - i) & 7][8 - j];
        Sre[sp][i][j] = re * w; Sim[sp][i][j] = im * w;
    }
    __syncthreads();
    { // inverse col DFT
        float re = 0.f, im = 0.f;
#pragma unroll
        for (int uu = 0; uu < 8; ++uu) {
            const int k = (uu * i) & 7;
            re += Sre[sp][uu][j] * FFT_CT[k] - Sim[sp][uu][j] * FFT_ST[k];
            im += Sre[sp][uu][j] * FFT_ST[k] + Sim[sp][uu][j] * FFT_CT[k];
        }
        Tre[sp][i][j] = re * 0.125f; Tim[sp][i][j] = im * 0.125f;
    }
    __syncthreads();
    { // inverse row DFT, real part, + residual
        float re = 0.f;
#pragma unroll
        for (int vv = 0; vv < 8; ++vv) {
            const int k = (vv * j) & 7;
            re += Tre[sp][i][vv] * FFT_CT[k] - Tim[sp][i][vv] * FFT_ST[k];
        }
        re *= 0.125f;
        const size_t addr = plane + yy * 256 + xx;
        out[addr] = xafter[addr] + re;
    }
}

// ---------------------------------------------------------------------------
// Host-side orchestration
// ---------------------------------------------------------------------------
// Workspace layout (lifetime-aliased):
static constexpr size_t OFF_A = 0;                       // qkv_pre/qkv -> t -> y2
static constexpr size_t SZ_A  = 268435456;               // 256 ch fp32
static constexpr size_t OFF_B = OFF_A + SZ_A;            // proj fp32 -> yT bf16
static constexpr size_t SZ_B  = 67108864;
static constexpr size_t OFF_C = OFF_B + SZ_B;            // x_after fp32
static constexpr size_t SZ_C  = 67108864;
static constexpr size_t OFF_D = OFF_C + SZ_C;            // ln1T -> resp -> ln2T
static constexpr size_t SZ_D  = 33554432;
static constexpr size_t OFF_E = OFF_D + SZ_D;            // projT bf16
static constexpr size_t SZ_E  = 33554432;
static constexpr size_t OFF_F = OFF_E + SZ_E;            // x_l fp32 -> fusedT bf16
static constexpr size_t SZ_F  = 33554432;
static constexpr size_t OFF_G = OFF_F + SZ_F;            // maskedT bf16
static constexpr size_t SZ_G  = 16777216;
static constexpr size_t OFF_H = OFF_G + SZ_G;            // small accumulators

extern "C" void kernel_launch(void* const* d_in, const int* in_sizes, int n_in,
                              void* d_out, int out_size, void* d_ws, size_t ws_size,
                              hipStream_t stream)
{
    (void)in_sizes; (void)n_in; (void)out_size; (void)ws_size;

    const float* x         = (const float*)d_in[0];
    const float* ln1_w     = (const float*)d_in[1];
    const float* ln1_b     = (const float*)d_in[2];
    const float* in_proj_w = (const float*)d_in[3];
    const float* qkv_w     = (const float*)d_in[4];
    const float* qkv_dw    = (const float*)d_in[5];
    const float* out_proj_w= (const float*)d_in[6];
    const float* temp      = (const float*)d_in[7];
    const float* scales    = (const float*)d_in[8];
    const float* gate_w1   = (const float*)d_in[9];
    const float* gate_b1   = (const float*)d_in[10];
    const float* gate_w2   = (const float*)d_in[11];
    const float* gate_b2   = (const float*)d_in[12];
    const float* pre_w     = (const float*)d_in[13];
    const float* pre_b     = (const float*)d_in[14];
    const float* post_w    = (const float*)d_in[15];
    const float* post_b    = (const float*)d_in[16];
    const float* m1w       = (const float*)d_in[17];
    const float* m1b       = (const float*)d_in[18];
    const float* m2w       = (const float*)d_in[19];
    const float* m2b       = (const float*)d_in[20];
    const float* spw       = (const float*)d_in[21];
    const float* spb       = (const float*)d_in[22];
    const float* ln2_w     = (const float*)d_in[23];
    const float* ln2_b     = (const float*)d_in[24];
    const float* ffn_in_w  = (const float*)d_in[25];
    const float* dw3_w     = (const float*)d_in[26];
    const float* dw5_w     = (const float*)d_in[27];
    const float* ffn_out_w = (const float*)d_in[28];
    const float* fft_W     = (const float*)d_in[29];

    char* ws = (char*)d_ws;
    float* qkv_pre = (float*)(ws + OFF_A);
    float* qkv     = (float*)(ws + OFF_A + 100663296);
    float* tbuf    = (float*)(ws + OFF_A);
    float* y2      = (float*)(ws + OFF_A);
    float* proj    = (float*)(ws + OFF_B);
    bf16*  yT      = (bf16*) (ws + OFF_B);
    float* x_after = (float*)(ws + OFF_C);
    bf16*  ln1T    = (bf16*) (ws + OFF_D);
    float* resp    = (float*)(ws + OFF_D);
    bf16*  ln2T    = (bf16*) (ws + OFF_D);
    bf16*  projT   = (bf16*) (ws + OFF_E);
    float* x_l     = (float*)(ws + OFF_F);
    bf16*  fusedT  = (bf16*) (ws + OFF_F);
    bf16*  maskedT = (bf16*) (ws + OFF_G);
    float* stats   = (float*)(ws + OFF_H);          // 16*80 floats
    float* thrsum  = (float*)(ws + OFF_H + 5120);   // 32 floats
    float* gsum    = (float*)(ws + OFF_H + 5248);   // 1 float
    float* attnf   = (float*)(ws + OFF_H + 5376);   // 16*64 floats
    float* cw      = (float*)(ws + OFF_H + 9472);   // 128 floats
    float* pooled  = (float*)(ws + OFF_H + 9984);   // 128 floats
    float* outp    = (float*)d_out;

    // zero atomic accumulators (stats + thrsum + gsum)
    hipMemsetAsync(ws + OFF_H, 0, 5264, stream);

    const int NT = NPIX / 16;   // 16384 pixel-tiles -> one block each

    // 1. LN1 -> bf16 pixel-major
    ln_kernel<<<1024, 256, 0, stream>>>(x, ln1_w, ln1_b, ln1T);

    // 2. in_proj (64x64) -> proj fp32 NCHW + projT bf16 [p][64]
    wmma_gemm_kernel<<<NT, 128, 0, stream>>>(ln1T, 64, 0, in_proj_w, nullptr,
        nullptr, 0, 0, proj, 64, 0, projT, 64, 0, 64, 64);

    // 3. LGCE pre (32x32)+bias -> x_l fp32
    wmma_gemm_kernel<<<NT, 64, 0, stream>>>(projT, 64, 0, pre_w, pre_b,
        nullptr, 0, 0, x_l, 32, 0, nullptr, 0, 0, 32, 32);

    // 4. CBAM channel path
    pool_kernel<<<128, 256, 0, stream>>>(x_l, pooled);
    cw_kernel<<<1, 128, 0, stream>>>(pooled, m1w, m1b, m2w, m2b, cw);
    resp_kernel<<<dim3(1024, 8), 256, 0, stream>>>(x_l, cw, spw, spb, resp, thrsum);
    mask_kernel<<<1024, 256, 0, stream>>>(x_l, resp, thrsum, maskedT);

    // 5. LGCE post (32x32)+bias+residual(local) -> fusedT channels 32..63
    wmma_gemm_kernel<<<NT, 64, 0, stream>>>(maskedT, 32, 0, post_w, post_b,
        proj, 64, 0, nullptr, 0, 0, fusedT, 64, 32, 32, 32);

    // 6. qkv 1x1 (96x32) -> qkv_pre fp32; then depthwise 3x3 -> qkv
    wmma_gemm_kernel<<<NT, 192, 0, stream>>>(projT, 64, 32, qkv_w, nullptr,
        nullptr, 0, 0, qkv_pre, 96, 0, nullptr, 0, 0, 96, 32);
    dwconv3_kernel<<<dim3(1024, 96), 256, 0, stream>>>(qkv_pre, qkv_dw, qkv, 96);

    // 7. gate mean (for dk) + attention statistics
    gate_mean_kernel<<<512, 256, 0, stream>>>(projT, gate_w1, gate_b1, gate_w2, gate_b2, gsum);
    qk_stats_kernel<<<dim3(32, 16), 256, 0, stream>>>(qkv, stats);
    attn_finalize_kernel<<<1, 128, 0, stream>>>(stats, gsum, temp, scales, attnf);
    attn_apply_kernel<<<dim3(256, 16), 256, 0, stream>>>(qkv, attnf, fusedT);

    // 8. out_proj (64x64) + residual x -> x_after fp32 NCHW
    wmma_gemm_kernel<<<NT, 128, 0, stream>>>(fusedT, 64, 0, out_proj_w, nullptr,
        x, 64, 0, x_after, 64, 0, nullptr, 0, 0, 64, 64);

    // 9. LN2 -> bf16 pixel-major
    ln_kernel<<<1024, 256, 0, stream>>>(x_after, ln2_w, ln2_b, ln2T);

    // 10. ffn_in (256x64) -> t fp32 NCHW
    wmma_gemm_kernel<<<NT, 512, 0, stream>>>(ln2T, 64, 0, ffn_in_w, nullptr,
        nullptr, 0, 0, tbuf, 256, 0, nullptr, 0, 0, 256, 64);

    // 11. fused dwconv3/dwconv5 + gated GELU -> yT bf16 [p][128]
    mffn_act_kernel<<<dim3(1024, 128), 256, 0, stream>>>(tbuf, dw3_w, dw5_w, yT);

    // 12. ffn_out (64x128) -> y2 fp32 NCHW
    wmma_gemm_kernel<<<NT, 128, 0, stream>>>(yT, 128, 0, ffn_out_w, nullptr,
        nullptr, 0, 0, y2, 64, 0, nullptr, 0, 0, 64, 128);

    // 13. per-patch FFT filter + residual -> final output
    fft_kernel<<<65536, 256, 0, stream>>>(y2, x_after, fft_W, outp);
}
